// SoftmaxMoELoRALinear_6468220747827
// MI455X (gfx1250) — matrix operand order repaired and checked by reading, not compiled, not run on hardware
//
#include <hip/hip_runtime.h>

// ---------------------------------------------------------------------------
// SoftmaxMoELoRALinear for MI455X (gfx1250, wave32, WMMA)
//   out = x @ W^T + SCALING * (gated LoRA)
// Strategy:
//   1) pack kernels: x -> bf16;  W -> wPack[K/32][N][32] bf16 (B-frag layout)
//      experts_B -> bfPack[4][2048][32];  [router|A] -> mPack[64][256][32]
//   2) GEMM1 (WMMA): Y(16384x256) = x_bf16 @ mPack   (logits + all-expert down)
//   3) gate: top-2 softmax over Y[:,0:8]; weighted(16384x128) bf16
//   4) GEMM2 (WMMA, fused): out = x_bf16 @ wPack  +  weighted @ bfPack
// Data movement: GLOBAL_LOAD_ASYNC_TO_LDS_B128 (ASYNCcnt), eliminating the
// VGPR staging that spilled to scratch in the ds_store pipeline.
// ---------------------------------------------------------------------------

typedef __bf16 bf16x8  __attribute__((ext_vector_type(8)));
typedef __bf16 bf16x16 __attribute__((ext_vector_type(16)));
typedef float  f32x8   __attribute__((ext_vector_type(8)));
typedef int    i32x4   __attribute__((ext_vector_type(4)));

#define AS1 __attribute__((address_space(1)))
#define AS3 __attribute__((address_space(3)))

#if __has_builtin(__builtin_amdgcn_global_load_async_to_lds_b128) && \
    __has_builtin(__builtin_amdgcn_s_wait_asynccnt)
#define USE_ASYNC_LDS 1
#else
#define USE_ASYNC_LDS 0
#endif

#define LDSS 40  // LDS row stride in halves (32 data + 8 pad), 80B: 16B aligned

static __device__ __forceinline__ unsigned short f2bf(float f) {
  union { float f; unsigned int u; } v; v.f = f;
  unsigned int r = v.u + 0x7FFFu + ((v.u >> 16) & 1u);  // round-to-nearest-even
  return (unsigned short)(r >> 16);
}

// --------------------------- pack kernels ----------------------------------

__global__ __launch_bounds__(256) void k_cvt_x(const float4* __restrict__ in,
                                               uint4* __restrict__ out) {
  size_t j = (size_t)blockIdx.x * 256 + threadIdx.x;   // one thread = 8 floats
  float4 a = in[2 * j], b = in[2 * j + 1];
  uint4 u;
  u.x = f2bf(a.x) | ((unsigned)f2bf(a.y) << 16);
  u.y = f2bf(a.z) | ((unsigned)f2bf(a.w) << 16);
  u.z = f2bf(b.x) | ((unsigned)f2bf(b.y) << 16);
  u.w = f2bf(b.z) | ((unsigned)f2bf(b.w) << 16);
  out[j] = u;
}

// wPack[kc][n][ki] = bf16(W[n][kc*32+ki]) ; W is (2048, 2048) row-major
__global__ __launch_bounds__(256) void k_pack_w(const float* __restrict__ w,
                                                unsigned short* __restrict__ wp) {
  int j = blockIdx.x * 256 + threadIdx.x;  // 131072 = 2048 n * 64 kc
  int n = j & 2047, kc = j >> 11;
  const float* src = w + (size_t)n * 2048 + kc * 32;
  unsigned int* dst = (unsigned int*)(wp + ((size_t)kc * 2048 + n) * 32);
#pragma unroll
  for (int i = 0; i < 32; i += 2)
    dst[i >> 1] = f2bf(src[i]) | ((unsigned)f2bf(src[i + 1]) << 16);
}

// bfPack[kc][n][ki] = bf16(Bflat[kc*32+ki][n]) ; Bflat = experts_B as (128,2048)
__global__ __launch_bounds__(256) void k_pack_bf(const float* __restrict__ eb,
                                                 unsigned short* __restrict__ bp) {
  int j = blockIdx.x * 256 + threadIdx.x;  // 8192 = 2048 n * 4 kc
  int n = j & 2047, kc = j >> 11;
  unsigned int* dst = (unsigned int*)(bp + ((size_t)kc * 2048 + n) * 32);
#pragma unroll
  for (int ki = 0; ki < 32; ki += 2) {
    float a = eb[(size_t)(kc * 32 + ki) * 2048 + n];
    float b = eb[(size_t)(kc * 32 + ki + 1) * 2048 + n];
    dst[ki >> 1] = f2bf(a) | ((unsigned)f2bf(b) << 16);
  }
}

// mPack[kc][n][ki]: n<8 router_w[n][k]; 16<=n<144 expert (n-16)/16 col (n-16)%16; else 0
__global__ __launch_bounds__(256) void k_pack_m(const float* __restrict__ rw,
                                                const float* __restrict__ ea,
                                                unsigned short* __restrict__ mp) {
  int j = blockIdx.x * 256 + threadIdx.x;  // 16384 = 256 n * 64 kc
  int n = j & 255, kc = j >> 8;
  unsigned int* dst = (unsigned int*)(mp + ((size_t)kc * 256 + n) * 32);
#pragma unroll
  for (int ki = 0; ki < 32; ki += 2) {
    float v0 = 0.f, v1 = 0.f;
    int k = kc * 32 + ki;
    if (n < 8) {
      v0 = rw[(size_t)n * 2048 + k];
      v1 = rw[(size_t)n * 2048 + k + 1];
    } else if (n >= 16 && n < 144) {
      int e = (n - 16) >> 4, r = (n - 16) & 15;
      v0 = ea[((size_t)e * 2048 + k) * 16 + r];
      v1 = ea[((size_t)e * 2048 + k + 1) * 16 + r];
    }
    dst[ki >> 1] = f2bf(v0) | ((unsigned)f2bf(v1) << 16);
  }
}

// --------------------------- gating ----------------------------------------
// Y row: [0:8] logits, [16+16e : 32+16e] expert-e down values
__global__ __launch_bounds__(256) void k_gate(const float* __restrict__ Y,
                                              unsigned short* __restrict__ Wt) {
  const float SC = 16.0f / 16.0f;  // ALPHA / RANK
  int t = blockIdx.x * 256 + threadIdx.x;
  const float* y = Y + (size_t)t * 256;
  float l[8];
#pragma unroll
  for (int e = 0; e < 8; e++) l[e] = y[e];
  int i0 = 0; float v0 = l[0];
#pragma unroll
  for (int e = 1; e < 8; e++) if (l[e] > v0) { v0 = l[e]; i0 = e; }
  int i1 = (i0 == 0) ? 1 : 0; float v1 = l[i1];
#pragma unroll
  for (int e = 0; e < 8; e++) if (e != i0 && l[e] > v1) { v1 = l[e]; i1 = e; }
  float ed = __expf(v1 - v0);
  float inv = 1.0f / (1.0f + ed);
  float g0 = SC * inv, g1 = SC * ed * inv;
  unsigned short* o = Wt + (size_t)t * 128;
#pragma unroll
  for (int e = 0; e < 8; e++) {
    float g = (e == i0) ? g0 : (e == i1) ? g1 : 0.0f;
#pragma unroll
    for (int r = 0; r < 16; r += 2) {
      unsigned int v = f2bf(g * y[16 + 16 * e + r]) |
                       ((unsigned)f2bf(g * y[16 + 16 * e + r + 1]) << 16);
      *(unsigned int*)(o + 16 * e + r) = v;
    }
  }
}

// --------------------------- WMMA GEMM -------------------------------------
// Out(M x Nfull) f32 = A1(bf16, lda1 B/row) @ B1pack  (+ A2 @ B2pack)
// Bpack layout: [kc][n][32] bf16.  Block tile 128x128, 8 waves of 32x64.
__global__ __launch_bounds__(256) void k_gemm_bf16(
    const char* __restrict__ A1, int lda1, const char* __restrict__ B1, int nIt1,
    const char* __restrict__ A2, int lda2, const char* __restrict__ B2, int nIt2,
    float* __restrict__ Out, int ldOut, int Nfull) {
  __shared__ __align__(16) unsigned short sA[2][128 * LDSS];
  __shared__ __align__(16) unsigned short sB[2][128 * LDSS];

  const int tid = threadIdx.x;
  const int lane = tid & 31;
  const int w = tid >> 5;
  const int wm = (w >> 1) * 32;   // wave M origin in tile (4 waves over M)
  const int wn = (w & 1) * 64;    // wave N origin in tile (2 waves over N)
  const int m0 = blockIdx.y * 128;
  const int n0 = blockIdx.x * 128;
  const int nTot = nIt1 + nIt2;

  f32x8 acc[2][4];
#pragma unroll
  for (int i = 0; i < 2; i++)
#pragma unroll
    for (int j = 0; j < 4; j++) acc[i][j] = (f32x8)(0.0f);

  // source selection for iteration `it`
  auto srcSel = [&](int it, const char*& A, size_t& lda, const char*& Bp, int& kc) {
    if (it < nIt1) { A = A1; lda = (size_t)lda1; Bp = B1; kc = it; }
    else           { A = A2; lda = (size_t)lda2; Bp = B2; kc = it - nIt1; }
  };

#if USE_ASYNC_LDS
  // --- global -> LDS via async DMA (no VGPR staging, tracked by ASYNCcnt) ---
  auto issueAsync = [&](int it, int buf) {
    const char* A; size_t lda; const char* Bp; int kc;
    srcSel(it, A, lda, Bp, kc);
#pragma unroll
    for (int i = 0; i < 2; i++) {
      int j = tid + 256 * i, r = j >> 2, s = j & 3;
      const char* ga = A + (size_t)(m0 + r) * lda + (size_t)kc * 64 + s * 16;
      const char* gb =
          Bp + ((size_t)kc * (size_t)Nfull + (size_t)(n0 + r)) * 64 + (size_t)s * 16;
      char* la = (char*)(&sA[buf][0]) + r * 80 + s * 16;
      char* lb = (char*)(&sB[buf][0]) + r * 80 + s * 16;
      __builtin_amdgcn_global_load_async_to_lds_b128((AS1 i32x4*)ga, (AS3 i32x4*)la,
                                                     0, 0);
      __builtin_amdgcn_global_load_async_to_lds_b128((AS1 i32x4*)gb, (AS3 i32x4*)lb,
                                                     0, 0);
    }
  };
#else
  // --- fallback: global -> VGPR -> ds_store pipeline ---
  uint4 ra[2], rb[2];
  auto loadG = [&](int it) {
    const char* A; size_t lda; const char* Bp; int kc;
    srcSel(it, A, lda, Bp, kc);
#pragma unroll
    for (int i = 0; i < 2; i++) {
      int j = tid + 256 * i, r = j >> 2, s = j & 3;
      ra[i] = *(const uint4*)(A + (size_t)(m0 + r) * lda + (size_t)kc * 64 + s * 16);
      rb[i] = *(const uint4*)(Bp + ((size_t)kc * (size_t)Nfull + (size_t)(n0 + r)) * 64 +
                              (size_t)s * 16);
    }
  };
  auto storeL = [&](int buf) {
#pragma unroll
    for (int i = 0; i < 2; i++) {
      int j = tid + 256 * i, r = j >> 2, s = j & 3;
      *(uint4*)((char*)(&sA[buf][0]) + r * 80 + s * 16) = ra[i];
      *(uint4*)((char*)(&sB[buf][0]) + r * 80 + s * 16) = rb[i];
    }
  };
#endif

  auto compute = [&](int buf) {
    const unsigned short* baseA = &sA[buf][0];
    const unsigned short* baseB = &sB[buf][0];
    const int lo16 = lane & 15, half = lane >> 4;
    bf16x16 af[2], bfv[4];
#pragma unroll
    for (int mt = 0; mt < 2; mt++) {
      // A 16x32 frag: lanes<16 row m K{0..7,16..23}; lanes>=16 K{8..15,24..31}
      const unsigned short* p = baseA + (wm + mt * 16 + lo16) * LDSS + half * 8;
      bf16x8 lo = *(const bf16x8*)p;
      bf16x8 hi = *(const bf16x8*)(p + 16);
      af[mt] = __builtin_shufflevector(lo, hi, 0, 1, 2, 3, 4, 5, 6, 7, 8, 9, 10,
                                       11, 12, 13, 14, 15);
    }
#pragma unroll
    for (int nt = 0; nt < 4; nt++) {
      // B 32x16 frag: lanes<16 col n K{0..7,16..23}; lanes>=16 K{8..15,24..31} (n-major LDS)
      const unsigned short* p = baseB + (wn + nt * 16 + lo16) * LDSS + half * 8;
      bf16x8 lo = *(const bf16x8*)p;
      bf16x8 hi = *(const bf16x8*)(p + 16);
      bfv[nt] = __builtin_shufflevector(lo, hi, 0, 1, 2, 3, 4, 5, 6, 7, 8, 9, 10,
                                        11, 12, 13, 14, 15);
    }
#pragma unroll
    for (int mt = 0; mt < 2; mt++)
#pragma unroll
      for (int nt = 0; nt < 4; nt++)
        acc[mt][nt] = __builtin_amdgcn_wmma_f32_16x16x32_bf16(
            false, af[mt], false, bfv[nt], (short)0, acc[mt][nt], false, false);
  };

#if USE_ASYNC_LDS
  issueAsync(0, 0);
  for (int it = 0; it < nTot; ++it) {
    __builtin_amdgcn_s_wait_asynccnt(0);  // my async writes to LDS are done
    __syncthreads();                      // everyone's are done
    if (it + 1 < nTot) issueAsync(it + 1, (it + 1) & 1);
    compute(it & 1);
  }
#else
  loadG(0);
  storeL(0);
  for (int it = 0; it < nTot; ++it) {
    __syncthreads();
    int buf = it & 1;
    if (it + 1 < nTot) loadG(it + 1);
    compute(buf);
    if (it + 1 < nTot) storeL(buf ^ 1);
  }
#endif

  // C/D layout: VGPR r -> M = r + (lane>=16 ? 8 : 0); N = lane & 15
  const int nlo = lane & 15, mh = (lane >> 4) * 8;
#pragma unroll
  for (int mt = 0; mt < 2; mt++)
#pragma unroll
    for (int nt = 0; nt < 4; nt++)
#pragma unroll
      for (int r = 0; r < 8; r++) {
        int mm = m0 + wm + mt * 16 + mh + r;
        int nn = n0 + wn + nt * 16 + nlo;
        Out[(size_t)mm * ldOut + nn] = acc[mt][nt][r];
      }
}

// --------------------------- launch ----------------------------------------

extern "C" void kernel_launch(void* const* d_in, const int* in_sizes, int n_in,
                              void* d_out, int out_size, void* d_ws, size_t ws_size,
                              hipStream_t stream) {
  (void)in_sizes; (void)n_in; (void)out_size; (void)ws_size;
  const float* x  = (const float*)d_in[0];  // (4,4096,2048)
  const float* wt = (const float*)d_in[1];  // (2048,2048)
  const float* rw = (const float*)d_in[2];  // (8,2048)
  const float* ea = (const float*)d_in[3];  // (8,2048,16)
  const float* eb = (const float*)d_in[4];  // (8,16,2048)

  char* ws = (char*)d_ws;
  unsigned short* xbf = (unsigned short*)(ws);                 // 64 MB
  unsigned short* wp  = (unsigned short*)(ws + 67108864);      //  8 MB
  unsigned short* bfp = (unsigned short*)(ws + 75497472);      // 0.5 MB
  unsigned short* mp  = (unsigned short*)(ws + 76021760);      //  1 MB
  float*          Y   = (float*)(ws + 77070336);               // 16 MB
  unsigned short* wgt = (unsigned short*)(ws + 93847552);      //  4 MB

  // 1) pack
  k_cvt_x<<<16384, 256, 0, stream>>>((const float4*)x, (uint4*)xbf);
  k_pack_w<<<512, 256, 0, stream>>>(wt, wp);
  k_pack_bf<<<32, 256, 0, stream>>>(eb, bfp);
  k_pack_m<<<64, 256, 0, stream>>>(rw, ea, mp);

  // 2) logits + all-expert LoRA-down: Y = x @ [router|A]  (N padded to 256)
  k_gemm_bf16<<<dim3(2, 128), 256, 0, stream>>>(
      (const char*)xbf, 4096, (const char*)mp, 64,
      (const char*)xbf, 4096, (const char*)mp, 0,
      Y, 256, 256);

  // 3) top-2 softmax gating -> weighted (16384 x 128) bf16
  k_gate<<<64, 256, 0, stream>>>(Y, wgt);

  // 4) fused: out = x @ W^T + weighted @ Bflat
  k_gemm_bf16<<<dim3(16, 128), 256, 0, stream>>>(
      (const char*)xbf, 4096, (const char*)wp, 64,
      (const char*)wgt, 256, (const char*)bfp, 4,
      (float*)d_out, 2048, 2048);
}
